// FeedForward_51719996178459
// MI455X (gfx1250) — compile-verified, matrix operand order checked by reading, not run on hardware
//
#include <hip/hip_runtime.h>
#include <hip/hip_bf16.h>
#include <math.h>

// ---------------- CDNA5 WMMA types ----------------
typedef __attribute__((ext_vector_type(16))) __bf16 v16bf;
typedef __attribute__((ext_vector_type(8)))  float  v8f;

__device__ __forceinline__ v8f wmma_bf16(v16bf a, v16bf b, v8f c) {
  // D(f32,16x16) = A(bf16,16x32) x B(bf16,32x16) + C
  return __builtin_amdgcn_wmma_f32_16x16x32_bf16(
      /*neg_a=*/false, a, /*neg_b=*/false, b,
      /*c_mod=*/(short)0, c, /*reuse_a=*/false, /*reuse_b=*/false);
}

// ---------------- problem constants ----------------
#define TKN   2048      // B*S tokens
#define DIM   1024      // D
#define FF    4096      // F
#define NE    8         // experts
#define NSLOT (2*TKN)   // every token routed to exactly 2 experts
#define LDSP  40        // LDS row stride (elements); 80B keeps 16B chunks aligned

// workspace header
struct Ws {
  int   counts[NE];
  int   fill[NE];
  int   offs[NE + 1];
  int   pad[7];
  int   tok_e[2 * TKN];     // per-token top-2 expert ids
  float tok_w[2 * TKN];     // per-token top-2 normalized weights
  int   slot_token[NSLOT];  // slot -> token
  float slot_w[NSLOT];      // slot -> routing weight
  int   tok_slot[2 * TKN];  // token -> its 2 slots
};

// ---------------- kernel 0: zero routing state ----------------
__global__ void k_zero(Ws* ws) {
  int i = threadIdx.x;
  if (i < NE) { ws->counts[i] = 0; ws->fill[i] = 0; }
}

// ---------------- kernel 1: gating (logits -> top2) ----------------
__global__ __launch_bounds__(256) void k_gate(const float* __restrict__ x,
                                              const float* __restrict__ gw,
                                              Ws* ws) {
  int t = blockIdx.x * blockDim.x + threadIdx.x;
  if (t >= TKN) return;
  float lg[NE];
#pragma unroll
  for (int e = 0; e < NE; ++e) lg[e] = 0.f;
  const float* xr = x + (size_t)t * DIM;
  for (int d = 0; d < DIM; ++d) {
    float xv = xr[d];
    const float* g = gw + (size_t)d * NE;
#pragma unroll
    for (int e = 0; e < NE; ++e) lg[e] = fmaf(xv, g[e], lg[e]);
  }
  int i0 = 0;
#pragma unroll
  for (int e = 1; e < NE; ++e) if (lg[e] > lg[i0]) i0 = e;
  int i1 = (i0 == 0) ? 1 : 0;
#pragma unroll
  for (int e = 0; e < NE; ++e) if (e != i0 && lg[e] > lg[i1]) i1 = e;
  float m  = fmaxf(lg[i0], lg[i1]);
  float e0 = __expf(lg[i0] - m), e1 = __expf(lg[i1] - m);
  float inv = 1.f / (e0 + e1);
  ws->tok_e[2 * t + 0] = i0; ws->tok_w[2 * t + 0] = e0 * inv;
  ws->tok_e[2 * t + 1] = i1; ws->tok_w[2 * t + 1] = e1 * inv;
  atomicAdd(&ws->counts[i0], 1);
  atomicAdd(&ws->counts[i1], 1);
}

// ---------------- kernel 2: prefix scan over 8 experts ----------------
__global__ void k_scan(Ws* ws) {
  int acc = 0;
  ws->offs[0] = 0;
  for (int e = 0; e < NE; ++e) { acc += ws->counts[e]; ws->offs[e + 1] = acc; }
}

// ---------------- kernel 3: scatter tokens into expert slots ----------------
__global__ __launch_bounds__(256) void k_scatter(Ws* ws) {
  int t = blockIdx.x * blockDim.x + threadIdx.x;
  if (t >= TKN) return;
#pragma unroll
  for (int k = 0; k < 2; ++k) {
    int e = ws->tok_e[2 * t + k];
    int pos = atomicAdd(&ws->fill[e], 1);
    int s = ws->offs[e] + pos;
    ws->slot_token[s] = t;
    ws->slot_w[s]     = ws->tok_w[2 * t + k];
    ws->tok_slot[2 * t + k] = s;
  }
}

// ---------------- stage-1 GEMM: h = silu(x@w1) * (x@w3)  (bf16 out) -------
// grid: (FF/128, TKN/64, NE), block 256 (8 waves).
// Block tile 64x128; wave tile M=64 x N=16 (4 A frags reuse each B frag).
__global__ __launch_bounds__(256) void k_ffn1(const float* __restrict__ x,
                                              const float* __restrict__ w1,
                                              const float* __restrict__ w3,
                                              const Ws* __restrict__ ws,
                                              __bf16* __restrict__ hbuf) {
  const int e     = blockIdx.z;
  const int base  = ws->offs[e];
  const int cnt   = ws->offs[e + 1] - base;
  const int mbase = blockIdx.y * 64;
  if (mbase >= cnt) return;
  const int fbase = blockIdx.x * 128;

  const int tid  = threadIdx.x;
  const int lane = tid & 31;
  const int wv   = tid >> 5;

  __shared__ __bf16 As[64][LDSP];   // 64 tokens x 32 K
  __shared__ int    toks[64];

  if (tid < 64) {
    int r = tid;
    toks[r] = (mbase + r < cnt) ? ws->slot_token[base + mbase + r] : -1;
  }
  __syncthreads();

  const int m    = lane & 15;
  const int half = lane >> 4;
  const int nn   = fbase + wv * 16 + (lane & 15);

  // 16-bit A/B fragment K-index pattern (ISA 7.12.2, 16-bit 16x32)
  int kidx[16];
#pragma unroll
  for (int j = 0; j < 16; ++j) {
    int r = j >> 1, o = j & 1;
    kidx[j] = ((r < 4) ? (2 * r) : (16 + 2 * (r - 4))) + (half ? 8 : 0) + o;
  }

  v8f c1[4] = {}, c3[4] = {};

  const int lr = tid >> 2;          // LDS fill: row (0..63)
  const int lk = (tid & 3) * 8;     // LDS fill: 8 cols

  for (int kb = 0; kb < DIM; kb += 32) {
    __syncthreads();
    {
      int tok = toks[lr];
      if (tok >= 0) {
        const size_t xo = (size_t)tok * DIM + kb + lk;
        const float4 v0 = *(const float4*)(x + xo);
        const float4 v1 = *(const float4*)(x + xo + 4);
        As[lr][lk + 0] = (__bf16)v0.x; As[lr][lk + 1] = (__bf16)v0.y;
        As[lr][lk + 2] = (__bf16)v0.z; As[lr][lk + 3] = (__bf16)v0.w;
        As[lr][lk + 4] = (__bf16)v1.x; As[lr][lk + 5] = (__bf16)v1.y;
        As[lr][lk + 6] = (__bf16)v1.z; As[lr][lk + 7] = (__bf16)v1.w;
      } else {
#pragma unroll
        for (int q = 0; q < 8; ++q) As[lr][lk + q] = (__bf16)0.f;
      }
    }
    __syncthreads();

    if (kb + 32 < DIM) {  // hint next weight K-block toward L2
      __builtin_prefetch(&w1[((size_t)e * DIM + kb + 32) * FF + nn], 0, 0);
      __builtin_prefetch(&w3[((size_t)e * DIM + kb + 32) * FF + nn], 0, 0);
    }

    v16bf a0, a1, a2, a3, b1, b3;
#pragma unroll
    for (int j = 0; j < 16; ++j) {
      const int k = kidx[j];
      a0[j] = As[m][k];
      a1[j] = As[m + 16][k];
      a2[j] = As[m + 32][k];
      a3[j] = As[m + 48][k];
      const size_t wof = ((size_t)e * DIM + (kb + k)) * FF + nn;
      b1[j] = (__bf16)w1[wof];
      b3[j] = (__bf16)w3[wof];
    }
    c1[0] = wmma_bf16(a0, b1, c1[0]);
    c1[1] = wmma_bf16(a1, b1, c1[1]);
    c1[2] = wmma_bf16(a2, b1, c1[2]);
    c1[3] = wmma_bf16(a3, b1, c1[3]);
    c3[0] = wmma_bf16(a0, b3, c3[0]);
    c3[1] = wmma_bf16(a1, b3, c3[1]);
    c3[2] = wmma_bf16(a2, b3, c3[2]);
    c3[3] = wmma_bf16(a3, b3, c3[3]);
  }

  // fused SwiGLU epilogue -> bf16 h
#pragma unroll
  for (int ti = 0; ti < 4; ++ti) {
#pragma unroll
    for (int j = 0; j < 8; ++j) {
      const int row = mbase + 16 * ti + j + (half ? 8 : 0);
      if (row < cnt) {
        float g  = c1[ti][j];
        float hv = (g / (1.f + __expf(-g))) * c3[ti][j];
        hbuf[(size_t)(base + row) * FF + nn] = (__bf16)hv;
      }
    }
  }
}

// ---------------- stage-2 GEMM: y = h @ w2  (f32 out, per slot) ----------
// grid: (DIM/128, TKN/64, NE), block 256 (8 waves). Wave tile M=64 x N=16.
// A tile staged with CDNA5 async global->LDS copy (ASYNCcnt-tracked).
__global__ __launch_bounds__(256) void k_ffn2(const __bf16* __restrict__ hbuf,
                                              const float* __restrict__ w2,
                                              const Ws* __restrict__ ws,
                                              float* __restrict__ ybuf) {
  const int e     = blockIdx.z;
  const int base  = ws->offs[e];
  const int cnt   = ws->offs[e + 1] - base;
  const int mbase = blockIdx.y * 64;
  if (mbase >= cnt) return;
  const int dbase = blockIdx.x * 128;

  const int tid  = threadIdx.x;
  const int lane = tid & 31;
  const int wv   = tid >> 5;

  __shared__ __bf16 As[64][LDSP];   // 64 slots x 32 K; rows 16B-chunk aligned

  const int m    = lane & 15;
  const int half = lane >> 4;
  const int nn   = dbase + wv * 16 + (lane & 15);

  int kidx[16];
#pragma unroll
  for (int j = 0; j < 16; ++j) {
    int r = j >> 1, o = j & 1;
    kidx[j] = ((r < 4) ? (2 * r) : (16 + 2 * (r - 4))) + (half ? 8 : 0) + o;
  }

  v8f ca[4] = {};

  const int lr = tid >> 2;          // staging row (0..63)
  const int lk = (tid & 3) * 8;     // staging chunk (8 bf16 = 16B)
  const bool rowok = (mbase + lr) < cnt;
  const __bf16* hrow = hbuf + (size_t)(base + mbase + lr) * FF;
  // LDS byte offset of this thread's staging destination (low 32 bits of the
  // generic address of a __shared__ object == LDS offset in the LDS aperture)
  const unsigned ldsoff = (unsigned)(size_t)&As[lr][lk];

  for (int kb = 0; kb < FF; kb += 32) {
    __syncthreads();   // previous fragment reads done before overwrite
    if (rowok) {
      unsigned long long g = (unsigned long long)(size_t)(hrow + kb + lk);
      // async 16B/lane copy global -> LDS, tracked by ASYNCcnt
      asm volatile("global_load_async_to_lds_b128 %0, %1, off"
                   :: "v"(ldsoff), "v"(g) : "memory");
    }
    asm volatile("s_wait_asynccnt 0x0" ::: "memory");
    __syncthreads();

    if (kb + 32 < FF)
      __builtin_prefetch(&w2[((size_t)e * FF + kb + 32) * DIM + nn], 0, 0);

    v16bf a0, a1, a2, a3, b2;
#pragma unroll
    for (int j = 0; j < 16; ++j) {
      const int k = kidx[j];
      a0[j] = As[m][k];
      a1[j] = As[m + 16][k];
      a2[j] = As[m + 32][k];
      a3[j] = As[m + 48][k];
      b2[j] = (__bf16)w2[((size_t)e * FF + (kb + k)) * DIM + nn];
    }
    ca[0] = wmma_bf16(a0, b2, ca[0]);
    ca[1] = wmma_bf16(a1, b2, ca[1]);
    ca[2] = wmma_bf16(a2, b2, ca[2]);
    ca[3] = wmma_bf16(a3, b2, ca[3]);
  }

#pragma unroll
  for (int ti = 0; ti < 4; ++ti) {
#pragma unroll
    for (int j = 0; j < 8; ++j) {
      const int row = mbase + 16 * ti + j + (half ? 8 : 0);
      if (row < cnt)
        ybuf[(size_t)(base + row) * DIM + nn] = ca[ti][j];
    }
  }
}

// ---------------- combine: out[t] = w0*y[s0] + w1*y[s1] ----------------
__global__ __launch_bounds__(256) void k_combine(const float* __restrict__ ybuf,
                                                 const Ws* __restrict__ ws,
                                                 float* __restrict__ out) {
  size_t idx = (size_t)blockIdx.x * blockDim.x + threadIdx.x;
  if (idx >= (size_t)TKN * DIM) return;
  int t = (int)(idx / DIM);
  int d = (int)(idx % DIM);
  int s0 = ws->tok_slot[2 * t + 0];
  int s1 = ws->tok_slot[2 * t + 1];
  out[idx] = ws->slot_w[s0] * ybuf[(size_t)s0 * DIM + d] +
             ws->slot_w[s1] * ybuf[(size_t)s1 * DIM + d];
}

// ---------------- launcher ----------------
extern "C" void kernel_launch(void* const* d_in, const int* in_sizes, int n_in,
                              void* d_out, int out_size, void* d_ws, size_t ws_size,
                              hipStream_t stream) {
  const float* x  = (const float*)d_in[0];  // [B,S,D]
  const float* gw = (const float*)d_in[1];  // [D,E]
  const float* w1 = (const float*)d_in[2];  // [E,D,F]
  const float* w2 = (const float*)d_in[3];  // [E,F,D]
  const float* w3 = (const float*)d_in[4];  // [E,D,F]
  float* out = (float*)d_out;

  Ws* ws = (Ws*)d_ws;
  size_t hdr = (sizeof(Ws) + 255) & ~(size_t)255;
  __bf16* hbuf = (__bf16*)((char*)d_ws + hdr);                         // 4096 x 4096 bf16
  float*  ybuf = (float*)((char*)d_ws + hdr + (size_t)NSLOT * FF * 2); // 4096 x 1024 f32

  k_zero<<<1, 32, 0, stream>>>(ws);
  k_gate<<<(TKN + 255) / 256, 256, 0, stream>>>(x, gw, ws);
  k_scan<<<1, 1, 0, stream>>>(ws);
  k_scatter<<<(TKN + 255) / 256, 256, 0, stream>>>(ws);

  dim3 g1(FF / 128, (TKN + 63) / 64, NE);
  k_ffn1<<<g1, 256, 0, stream>>>(x, w1, w3, ws, hbuf);

  dim3 g2(DIM / 128, (TKN + 63) / 64, NE);
  k_ffn2<<<g2, 256, 0, stream>>>(hbuf, w2, ws, ybuf);

  size_t nelt = (size_t)TKN * DIM;
  k_combine<<<(unsigned)((nelt + 255) / 256), 256, 0, stream>>>(ybuf, ws, out);
}
